// SelfSelfAttention_6442450944638
// MI455X (gfx1250) — matrix operand order repaired and checked by reading, not compiled, not run on hardware
//
#include <hip/hip_runtime.h>

// ---------------------------------------------------------------------------
// Types for CDNA5 WMMA (wave32): 16x16x32 bf16 -> f32 accumulate
// ---------------------------------------------------------------------------
typedef __attribute__((ext_vector_type(16))) __bf16 v16bf;
typedef __attribute__((ext_vector_type(8)))  float  v8f;

union FragB {
  v16bf v;
  unsigned short s[16];
  uint4 q[2];
};

__device__ __forceinline__ unsigned short f2bf(float f) {
  union { float f; unsigned int u; } x; x.f = f;
  unsigned int u = x.u;
  u += 0x7FFFu + ((u >> 16) & 1u);           // round-to-nearest-even
  return (unsigned short)(u >> 16);
}
__device__ __forceinline__ float bf2f(unsigned short h) {
  union { unsigned int u; float f; } x; x.u = ((unsigned int)h) << 16;
  return x.f;
}

// A-matrix 16x32 bf16 fragment (ISA: lanes 0-15 row M=lane, K=0..7,16..23;
// lanes 16-31 same row, K=8..15,24..31). Caller passes koff = k0 + half*8.
__device__ __forceinline__ v16bf load_fragA(const unsigned short* p, int koff) {
  FragB f;
  f.q[0] = *(const uint4*)(p + koff);
  f.q[1] = *(const uint4*)(p + koff + 16);
  return f.v;
}
// B-matrix 32x16 bf16 fragment (same K order as A: lanes 0-15 K=0..7,16..23,
// lanes 16-31 K=8..15,24..31). p = start of the K-contiguous column.
__device__ __forceinline__ v16bf load_fragB(const unsigned short* p, int k0, int half) {
  FragB f;
  const unsigned short* q = p + k0 + half * 8;
  f.q[0] = *(const uint4*)(q);
  f.q[1] = *(const uint4*)(q + 16);
  return f.v;
}

__device__ __forceinline__ v8f wmma_bf16(v16bf a, v16bf b, v8f c) {
  return __builtin_amdgcn_wmma_f32_16x16x32_bf16(false, a, false, b, (short)0, c,
                                                 false, false);
}

__device__ __forceinline__ float redmax16(float v) {
  v = fmaxf(v, __shfl_xor(v, 8, 32));
  v = fmaxf(v, __shfl_xor(v, 4, 32));
  v = fmaxf(v, __shfl_xor(v, 2, 32));
  v = fmaxf(v, __shfl_xor(v, 1, 32));
  return v;
}
__device__ __forceinline__ float redsum16(float v) {
  v += __shfl_xor(v, 8, 32);
  v += __shfl_xor(v, 4, 32);
  v += __shfl_xor(v, 2, 32);
  v += __shfl_xor(v, 1, 32);
  return v;
}

// ---------------------------------------------------------------------------
// Elementwise / reduction kernels
// ---------------------------------------------------------------------------
__global__ void cvt_x_kernel(const float* __restrict__ x,
                             unsigned short* __restrict__ xb, int Ntok) {
  size_t i = (size_t)blockIdx.x * blockDim.x + threadIdx.x;
  size_t total = (size_t)Ntok * 8 * 768;
  if (i >= total) return;
  int c = (int)(i % 768);
  size_t bn = i / 768;
  int n = (int)(bn % (size_t)Ntok);
  int b = (int)(bn / (size_t)Ntok);
  xb[i] = f2bf(x[((size_t)n * 8 + b) * 768 + c]);   // x is [N,B,C]
}

__global__ void cvt_f32_bf16_kernel(const float* __restrict__ in,
                                    unsigned short* __restrict__ out,
                                    size_t n, float alpha) {
  size_t i = (size_t)blockIdx.x * blockDim.x + threadIdx.x;
  if (i >= n) return;
  out[i] = f2bf(in[i] * alpha);
}

__global__ void inv_temp_kernel(const float* __restrict__ x,
                                float* __restrict__ itemp, int Ntok) {
  int b = blockIdx.x;
  int tid = threadIdx.x;
  float acc = 0.f;
  for (int n = tid; n < Ntok; n += blockDim.x) {
    const float* p = x + ((size_t)n * 8 + b) * 768;
    float s = 0.f;
    for (int c = 0; c < 768; ++c) { float v = p[c]; s += v * v; }
    acc += sqrtf(s);
  }
  __shared__ float red[256];
  red[tid] = acc;
  __syncthreads();
  for (int off = 128; off > 0; off >>= 1) {
    if (tid < off) red[tid] += red[tid + off];
    __syncthreads();
  }
  if (tid == 0) itemp[b] = (red[0] / (float)Ntok) * 0.125f;  // * hd^-0.5
}

__global__ void l2norm_kernel(const unsigned short* __restrict__ in,
                              unsigned short* __restrict__ out, int rows) {
  int r = blockIdx.x * blockDim.x + threadIdx.x;
  if (r >= rows) return;
  const unsigned short* p = in + (size_t)r * 64;
  unsigned short* q = out + (size_t)r * 64;
  float v[64];
  float s = 0.f;
#pragma unroll
  for (int i = 0; i < 64; ++i) { v[i] = bf2f(p[i]); s += v[i] * v[i]; }
  float inv = 1.0f / fmaxf(sqrtf(s), 1e-12f);
#pragma unroll
  for (int i = 0; i < 64; ++i) q[i] = f2bf(v[i] * inv);
}

// [BH, Ntok, 64] -> [BH, 64, Np] (zero padded tail so wmma sees 0*p, not NaN)
__global__ void transpose_hd_kernel(const unsigned short* __restrict__ in,
                                    unsigned short* __restrict__ out,
                                    int Ntok, int Np) {
  size_t i = (size_t)blockIdx.x * blockDim.x + threadIdx.x;
  size_t total = (size_t)96 * 64 * Np;
  if (i >= total) return;
  int n = (int)(i % (size_t)Np);
  size_t r = i / (size_t)Np;
  int d = (int)(r % 64);
  size_t bh = r / 64;
  out[i] = (n < Ntok) ? in[(bh * (size_t)Ntok + n) * 64 + d] : (unsigned short)0;
}

// ---------------------------------------------------------------------------
// bf16 GEMM: out[M x Nout] = A[M x K] * W[Nout x K]^T  (both K-contiguous)
// 8 waves/block, each wave -> 16(M) x 64(N) tile, 4 accumulators.
// k-loop unrolled x2 with explicit ping-pong fragment sets: each set has a
// single static def and use site, so double-buffering costs zero v_mov -- the
// prefetch loads retire under the other set's 4 wmmas. Requires K % 64 == 0.
// mode 0: scatter into q/k/v bf16 [B,H,N,64] (Nout = 2304)
// mode 1: f32 + bias into [N,B,C] layout (Nout = 768)
// ---------------------------------------------------------------------------
__global__ void gemm_bf16_kernel(const unsigned short* __restrict__ A,
                                 const unsigned short* __restrict__ W,
                                 const float* __restrict__ bias,
                                 unsigned short* __restrict__ oq,
                                 unsigned short* __restrict__ ok,
                                 unsigned short* __restrict__ ov,
                                 float* __restrict__ of,
                                 int M, int K, int mode, int Ntok) {
  const int lane = threadIdx.x & 31;
  const int wid  = threadIdx.x >> 5;
  const int half = lane >> 4;
  const int lc   = lane & 15;
  const int m0   = (blockIdx.x * 8 + wid) * 16;
  const int n0   = blockIdx.y * 64;

  v8f z = {0.f, 0.f, 0.f, 0.f, 0.f, 0.f, 0.f, 0.f};
  v8f acc[4] = {z, z, z, z};

  int arow = m0 + lc;
  if (arow > M - 1) arow = M - 1;
  const unsigned short* Ar = A + (size_t)arow * K;
  const unsigned short* W0 = W + (size_t)(n0 + lc) * K;
  const unsigned short* W1 = W0 + (size_t)16 * K;
  const unsigned short* W2 = W0 + (size_t)32 * K;
  const unsigned short* W3 = W0 + (size_t)48 * K;

  // ping-pong set 0: fragments for k0 = 0
  v16bf a0 = load_fragA(Ar, half * 8);
  v16bf b00 = load_fragB(W0, 0, half);
  v16bf b01 = load_fragB(W1, 0, half);
  v16bf b02 = load_fragB(W2, 0, half);
  v16bf b03 = load_fragB(W3, 0, half);

  for (int k0 = 0; k0 < K; k0 += 64) {
    // prefetch set 1 (k0+32, always < K since K % 64 == 0)
    int k1 = k0 + 32;
    v16bf a1  = load_fragA(Ar, k1 + half * 8);
    v16bf b10 = load_fragB(W0, k1, half);
    v16bf b11 = load_fragB(W1, k1, half);
    v16bf b12 = load_fragB(W2, k1, half);
    v16bf b13 = load_fragB(W3, k1, half);

    acc[0] = wmma_bf16(a0, b00, acc[0]);
    acc[1] = wmma_bf16(a0, b01, acc[1]);
    acc[2] = wmma_bf16(a0, b02, acc[2]);
    acc[3] = wmma_bf16(a0, b03, acc[3]);

    // prefetch set 0 for the next unrolled iteration (k0+64)
    int k2 = k0 + 64;
    if (k2 < K) {
      a0  = load_fragA(Ar, k2 + half * 8);
      b00 = load_fragB(W0, k2, half);
      b01 = load_fragB(W1, k2, half);
      b02 = load_fragB(W2, k2, half);
      b03 = load_fragB(W3, k2, half);
    }

    acc[0] = wmma_bf16(a1, b10, acc[0]);
    acc[1] = wmma_bf16(a1, b11, acc[1]);
    acc[2] = wmma_bf16(a1, b12, acc[2]);
    acc[3] = wmma_bf16(a1, b13, acc[3]);
  }

#pragma unroll
  for (int j = 0; j < 4; ++j) {
#pragma unroll
    for (int i = 0; i < 8; ++i) {
      int row = m0 + i + 8 * half;      // C-layout: lanes 0-15 row i, 16-31 row i+8
      if (row >= M) continue;
      int col = n0 + j * 16 + lc;
      float val = acc[j][i];
      int b = row / Ntok, n = row % Ntok;
      if (mode == 0) {
        int s3 = col / 768;
        int rem = col % 768;
        int h = rem >> 6, d = rem & 63;
        unsigned short* dst = (s3 == 0) ? oq : (s3 == 1) ? ok : ov;
        dst[(((size_t)b * 12 + h) * Ntok + n) * 64 + d] = f2bf(val);
      } else {
        of[((size_t)n * 8 + b) * 768 + col] = val + bias[col];
      }
    }
  }
}

// ---------------------------------------------------------------------------
// Flash attention, one wave per 16-row Q tile, 4 waves/block.
// Q,K: bf16 [B,H,N,64]; Vt: bf16 [B,H,64,Np] (pre-transposed, zero-padded).
// V fragments are fetched BEFORE the softmax VALU block so global latency is
// hidden under the exp/shuffle work.
// mode 0: bf16 [B,H,N,64] out; mode 1: bf16 [B,N,C] out;
// mode 2: f32 [B,N,C] out, out = alpha*attn + (beta ? out : 0)
// ---------------------------------------------------------------------------
__global__ void flash_kernel(const unsigned short* __restrict__ Q,
                             const unsigned short* __restrict__ Kr,
                             const unsigned short* __restrict__ Vt,
                             const float* __restrict__ scale_arr, float scale_const,
                             unsigned short* __restrict__ out_bf,
                             float* __restrict__ out_f,
                             int mode, float alpha, int beta,
                             int Ntok, int Np) {
  const int lane = threadIdx.x & 31;
  const int wid  = threadIdx.x >> 5;
  const int half = lane >> 4;
  const int lc   = lane & 15;
  const int h    = blockIdx.y;
  const int b    = blockIdx.z;
  const int bh   = b * 12 + h;
  const int r0   = (blockIdx.x * 4 + wid) * 16;

  int qrow = r0 + lc;
  if (qrow > Ntok - 1) qrow = Ntok - 1;
  const unsigned short* qp = Q + ((size_t)bh * Ntok + qrow) * 64;
  v16bf aq0 = load_fragA(qp, half * 8);
  v16bf aq1 = load_fragA(qp, 32 + half * 8);

  const float temp = scale_arr ? scale_arr[b] : scale_const;
  const unsigned short* kb_ = Kr + (size_t)bh * Ntok * 64;
  const unsigned short* vb_ = Vt + (size_t)bh * 64 * Np;
  const unsigned short* vrow0 = vb_ + (size_t)(lc) * Np;
  const unsigned short* vrow1 = vb_ + (size_t)(16 + lc) * Np;
  const unsigned short* vrow2 = vb_ + (size_t)(32 + lc) * Np;
  const unsigned short* vrow3 = vb_ + (size_t)(48 + lc) * Np;

  v8f z = {0.f, 0.f, 0.f, 0.f, 0.f, 0.f, 0.f, 0.f};
  v8f o0 = z, o1 = z, o2 = z, o3 = z;
  float rmax[8], rsum[8];
#pragma unroll
  for (int i = 0; i < 8; ++i) { rmax[i] = -3.0e38f; rsum[i] = 0.f; }

  __shared__ __align__(16) unsigned short Pl[4][16][32];
  unsigned short (*P)[32] = Pl[wid];

  for (int j0 = 0; j0 < Ntok; j0 += 32) {
    // ---- S = Q K^T for two 16-col subtiles ----
    v8f s0 = z, s1 = z;
    {
      int col = j0 + lc;
      int cc = (col < Ntok) ? col : (Ntok - 1);
      const unsigned short* kp = kb_ + (size_t)cc * 64;
      s0 = wmma_bf16(aq0, load_fragB(kp, 0, half), s0);
      s0 = wmma_bf16(aq1, load_fragB(kp, 32, half), s0);
    }
    {
      int col = j0 + 16 + lc;
      int cc = (col < Ntok) ? col : (Ntok - 1);
      const unsigned short* kp = kb_ + (size_t)cc * 64;
      s1 = wmma_bf16(aq0, load_fragB(kp, 0, half), s1);
      s1 = wmma_bf16(aq1, load_fragB(kp, 32, half), s1);
    }

    // ---- issue V fragment loads now; they retire under the softmax VALU ----
    v16bf bv0 = load_fragB(vrow0, j0, half);
    v16bf bv1 = load_fragB(vrow1, j0, half);
    v16bf bv2 = load_fragB(vrow2, j0, half);
    v16bf bv3 = load_fragB(vrow3, j0, half);

    const bool cv0 = (j0 + lc) < Ntok;
    const bool cv1 = (j0 + 16 + lc) < Ntok;

    // ---- online softmax (row m = i + 8*half lives across a 16-lane half) ----
#pragma unroll
    for (int i = 0; i < 8; ++i) {
      float a0 = cv0 ? s0[i] * temp : -3.0e38f;
      float a1 = cv1 ? s1[i] * temp : -3.0e38f;
      float mx = fmaxf(redmax16(a0), redmax16(a1));
      float nm = fmaxf(rmax[i], mx);
      float p0 = __expf(a0 - nm);
      float p1 = __expf(a1 - nm);
      float ps = redsum16(p0) + redsum16(p1);
      float cr = __expf(rmax[i] - nm);
      rsum[i] = rsum[i] * cr + ps;
      rmax[i] = nm;
      o0[i] *= cr; o1[i] *= cr; o2[i] *= cr; o3[i] *= cr;
      P[i + 8 * half][lc]      = f2bf(p0);
      P[i + 8 * half][16 + lc] = f2bf(p1);
    }
    __syncthreads();  // fence LDS writes before cross-lane A-layout reads

    // ---- P (C-layout) -> A-layout via LDS, then O += P * V ----
    FragB ap;
    ap.q[0] = *(const uint4*)&P[lc][half * 8];
    ap.q[1] = *(const uint4*)&P[lc][half * 8 + 16];
    o0 = wmma_bf16(ap.v, bv0, o0);
    o1 = wmma_bf16(ap.v, bv1, o1);
    o2 = wmma_bf16(ap.v, bv2, o2);
    o3 = wmma_bf16(ap.v, bv3, o3);
  }

#pragma unroll
  for (int i = 0; i < 8; ++i) {
    int grow = r0 + i + 8 * half;
    if (grow >= Ntok) continue;
    float inv = 1.0f / rsum[i];
    float vals[4] = {o0[i] * inv, o1[i] * inv, o2[i] * inv, o3[i] * inv};
#pragma unroll
    for (int c = 0; c < 4; ++c) {
      int col = c * 16 + lc;
      if (mode == 0) {
        out_bf[((size_t)bh * Ntok + grow) * 64 + col] = f2bf(vals[c]);
      } else if (mode == 1) {
        out_bf[((size_t)b * Ntok + grow) * 768 + h * 64 + col] = f2bf(vals[c]);
      } else {
        size_t oi = ((size_t)b * Ntok + grow) * 768 + h * 64 + col;
        float r = vals[c] * alpha;
        if (beta) r += out_f[oi];
        out_f[oi] = r;
      }
    }
  }
}

// ---------------------------------------------------------------------------
// Host-side orchestration
// ---------------------------------------------------------------------------
extern "C" void kernel_launch(void* const* d_in, const int* in_sizes, int n_in,
                              void* d_out, int out_size, void* d_ws, size_t ws_size,
                              hipStream_t stream) {
  (void)n_in; (void)out_size; (void)ws_size;
  const float* x     = (const float*)d_in[0];
  const float* Wqkv  = (const float*)d_in[1];
  const float* Wproj = (const float*)d_in[2];
  const float* bproj = (const float*)d_in[3];
  float* out = (float*)d_out;

  const int B = 8, C = 768, H = 12;
  const int Ntok = in_sizes[0] / (B * C);               // 1569
  const int Np   = ((Ntok + 31) / 32) * 32;             // padded kv length
  const int MALL = B * Ntok;                            // 12552
  const size_t NBC = (size_t)MALL * C;

  char* base = (char*)d_ws;
  auto alloc = [&](size_t bytes) -> char* {
    char* p = base;
    base += (bytes + 255) & ~(size_t)255;
    return p;
  };
  unsigned short* xb   = (unsigned short*)alloc(NBC * 2);
  unsigned short* wqkv = (unsigned short*)alloc((size_t)3 * C * C * 2);
  unsigned short* wprj = (unsigned short*)alloc((size_t)C * C * 2);
  unsigned short* qb   = (unsigned short*)alloc(NBC * 2);
  unsigned short* kbuf = (unsigned short*)alloc(NBC * 2);
  unsigned short* vbuf = (unsigned short*)alloc(NBC * 2);
  unsigned short* vt   = (unsigned short*)alloc((size_t)B * H * 64 * Np * 2);
  unsigned short* nbuf = (unsigned short*)alloc(NBC * 2);
  unsigned short* tbuf = (unsigned short*)alloc((size_t)B * H * 64 * Np * 2);
  unsigned short* ybuf = (unsigned short*)alloc(NBC * 2);
  unsigned short* pin  = (unsigned short*)alloc(NBC * 2);
  float* ssacc = (float*)alloc(NBC * 4);
  float* itemp = (float*)alloc(64);

  dim3 b256(256), b128(128);

  // 1) conversions + inv_temp
  cvt_x_kernel<<<dim3((unsigned)((NBC + 255) / 256)), b256, 0, stream>>>(x, xb, Ntok);
  cvt_f32_bf16_kernel<<<dim3((unsigned)(((size_t)3 * C * C + 255) / 256)), b256, 0, stream>>>(
      Wqkv, wqkv, (size_t)3 * C * C, 1.0f);
  cvt_f32_bf16_kernel<<<dim3((unsigned)(((size_t)C * C + 255) / 256)), b256, 0, stream>>>(
      Wproj, wprj, (size_t)C * C, 1.0f);
  inv_temp_kernel<<<dim3(B), b256, 0, stream>>>(x, itemp, Ntok);

  // 2) qkv GEMM (bf16 WMMA) -> q,k,v [B,H,N,64]
  dim3 gq((MALL + 127) / 128, (3 * C) / 64);
  gemm_bf16_kernel<<<gq, b256, 0, stream>>>(xb, wqkv, nullptr, qb, kbuf, vbuf,
                                            nullptr, MALL, C, 0, Ntok);

  // 3) V^T for the P*V WMMA B-operand
  size_t ttot = (size_t)B * H * 64 * Np;
  transpose_hd_kernel<<<dim3((unsigned)((ttot + 255) / 256)), b256, 0, stream>>>(
      vbuf, vt, Ntok, Np);

  // 4) original attention -> pin(bf16 [B,N,C]) -> proj -> x_ori (second output)
  int qtiles = (Ntok + 15) / 16;
  dim3 gf((qtiles + 3) / 4, H, B);
  flash_kernel<<<gf, b128, 0, stream>>>(qb, kbuf, vt, nullptr, 0.125f,
                                        pin, nullptr, 1, 1.0f, 0, Ntok, Np);
  dim3 gp((MALL + 127) / 128, C / 64);
  gemm_bf16_kernel<<<gp, b256, 0, stream>>>(pin, wprj, bproj, nullptr, nullptr,
                                            nullptr, out + NBC, MALL, C, 1, Ntok);

  // 5) three self-self branches, accumulated *1/3 into ssacc (f32)
  const unsigned short* srcs[3] = {vbuf, kbuf, qb};
  int rows = B * H * Ntok;
  dim3 gl((rows + 255) / 256);
  for (int it = 0; it < 3; ++it) {
    l2norm_kernel<<<gl, b256, 0, stream>>>(srcs[it], nbuf, rows);
    transpose_hd_kernel<<<dim3((unsigned)((ttot + 255) / 256)), b256, 0, stream>>>(
        nbuf, tbuf, Ntok, Np);
    // step 1: y = attn(xn, xn, xn) with inv_temp
    flash_kernel<<<gf, b128, 0, stream>>>(nbuf, nbuf, tbuf, itemp, 0.f,
                                          ybuf, nullptr, 0, 1.0f, 0, Ntok, Np);
    // step 2: out += 1/3 * attn(l2norm(y), l2norm(y), v)
    l2norm_kernel<<<gl, b256, 0, stream>>>(ybuf, nbuf, rows);
    flash_kernel<<<gf, b128, 0, stream>>>(nbuf, nbuf, vt, itemp, 0.f,
                                          nullptr, ssacc, 2, (1.0f / 3.0f),
                                          (it > 0) ? 1 : 0, Ntok, Np);
  }

  // 6) project accumulated ss output -> x_out (first output)
  cvt_f32_bf16_kernel<<<dim3((unsigned)((NBC + 255) / 256)), b256, 0, stream>>>(
      ssacc, pin, NBC, 1.0f);
  gemm_bf16_kernel<<<gp, b256, 0, stream>>>(pin, wprj, bproj, nullptr, nullptr,
                                            nullptr, out, MALL, C, 1, Ntok);
}
